// RelativeMultiHeadAttention_84378927497357
// MI455X (gfx1250) — matrix operand order repaired and checked by reading, not compiled, not run on hardware
//
#include <hip/hip_runtime.h>

typedef _Float16 h16;
typedef __attribute__((ext_vector_type(16))) _Float16 v16h;
typedef __attribute__((ext_vector_type(8)))  float    v8f;
typedef __attribute__((ext_vector_type(4)))  unsigned int u32x4;
typedef __attribute__((ext_vector_type(4)))  int i32x4;
typedef __attribute__((ext_vector_type(8)))  int i32x8;

union frag16 { v16h v; uint4 q[2]; };
union pack8  { uint4 q; h16 e[8]; };
union pack64 { uint4 q[8]; h16 e[64]; };

#define WMMA(a, b, c) __builtin_amdgcn_wmma_f32_16x16x32_f16(false, (a), false, (b), (short)0, (c), false, false)

#define USE_TDM 1

#define S_LEN  1024
#define DMODEL 512
#define DH     64
#define NROWS  8192            // B * S
#define SCALE  0.044194173824159216f   // 1/sqrt(512)

// ---------------------------------------------------------------------------
// f32 -> f16 elementwise convert (4 elements / thread)
// ---------------------------------------------------------------------------
__global__ void f32_to_f16_kernel(const float* __restrict__ src, h16* __restrict__ dst, int n4) {
    int i = blockIdx.x * blockDim.x + threadIdx.x;
    if (i < n4) {
        float4 v = ((const float4*)src)[i];
        pack8 p;
        p.e[0] = (h16)v.x; p.e[1] = (h16)v.y; p.e[2] = (h16)v.z; p.e[3] = (h16)v.w;
        ((uint2*)dst)[i] = make_uint2(p.q.x, p.q.y);
    }
}

// ---------------------------------------------------------------------------
// Tiled WMMA GEMM:  C[M,N] = A16[M,K] @ W16[K,N] (+ bias[col]) with f32 accum.
//   outF != null : write f32 (final projection -> d_out)
//   else         : out0 = f16(acc+bias+add0[col]); if out1: f16(acc+bias+add1[col])
// Block tile 128(M) x 64(N), K-step 32, 256 threads = 8 waves.
// A tile staged by the Tensor Data Mover (wave 0 issues, TENSORcnt + barrier).
// B tile stored transposed in LDS so B fragments are wide ds loads.
// ---------------------------------------------------------------------------
#define TM 128
#define TN 64
#define TK 32

__global__ __launch_bounds__(256) void gemm16_kernel(
    const h16* __restrict__ A, const h16* __restrict__ W,
    const float* __restrict__ bias, const float* __restrict__ add0,
    const float* __restrict__ add1,
    h16* __restrict__ out0, h16* __restrict__ out1, float* __restrict__ outF,
    int M, int N, int K)
{
    __shared__ __align__(16) h16 As[TM][TK + 8];    // row stride 80B
    __shared__ __align__(16) h16 BsT[TN][TK + 24];  // transposed, row stride 112B

    const int t = threadIdx.x;
    const int w = t >> 5, lane = t & 31, nl = lane & 15, half = lane >> 4;
    const int m0 = blockIdx.y * TM, n0 = blockIdx.x * TN;

    v8f z = {};
    v8f acc[4];
#pragma unroll
    for (int i = 0; i < 4; ++i) acc[i] = z;

    const int arow_ld = t >> 1, acol_ld = (t & 1) * 16;  // fallback A loader
    const int brow_ld = t >> 3, bcol_ld = (t & 7) * 8;   // B: 8 thr/row, 8 h16 each

    for (int k0 = 0; k0 < K; k0 += TK) {
        __syncthreads();
#if USE_TDM
        if (t < 32) {   // wave 0 issues one TDM descriptor for the whole A tile
            unsigned long long ga = (unsigned long long)(const void*)(A + (size_t)m0 * K + k0);
            unsigned ldsa = (unsigned)(unsigned long long)(const void*)&As[0][0];
            u32x4 g0;
            g0[0] = 1u;                                   // count=1, no gather
            g0[1] = ldsa;                                 // lds_addr
            g0[2] = (unsigned)(ga & 0xFFFFFFFFu);         // global_addr lo
            g0[3] = (unsigned)((ga >> 32) & 0x1FFFFFFu) | 0x80000000u; // hi | type=2
            i32x8 g1;
            g1[0] = (1 << 16)      // data_size = 2 bytes
                  | (1 << 20)      // pad_enable
                  | (3 << 22)      // pad_interval = 16 DWORDs (one 64B row)
                  | (3 << 25);     // pad_amount   = 4 DWORDs (8 h16)
            g1[1] = (int)((unsigned)TK << 16);    // tensor_dim0 = 32
            g1[2] = (int)((unsigned)TM << 16);    // tensor_dim1 = 128
            g1[3] = (int)((unsigned)TK << 16);    // tile_dim0 = 32
            g1[4] = TM;                           // tile_dim1 = 128, tile_dim2 = 0
            g1[5] = K;                            // tensor_dim0_stride = K elements
            g1[6] = 0; g1[7] = 0;
            i32x4 zz = {0, 0, 0, 0};
#if __clang_major__ >= 23
            i32x8 z8 = {0, 0, 0, 0, 0, 0, 0, 0};
            __builtin_amdgcn_tensor_load_to_lds(g0, g1, zz, zz, z8, 0);
#else
            __builtin_amdgcn_tensor_load_to_lds(g0, g1, zz, zz, 0);
#endif
        }
#else
        {
            const uint4* s = (const uint4*)(A + (size_t)(m0 + arow_ld) * K + k0 + acol_ld);
            uint4* d = (uint4*)&As[arow_ld][acol_ld];
            d[0] = s[0]; d[1] = s[1];
        }
#endif
        {   // B tile: coalesced row read, transpose-scatter into BsT[n][k]
            pack8 wv;
            wv.q = *(const uint4*)(W + (size_t)(k0 + brow_ld) * N + n0 + bcol_ld);
#pragma unroll
            for (int i = 0; i < 8; ++i) BsT[bcol_ld + i][brow_ld] = wv.e[i];
        }
        if (k0 + TK < K)
            __builtin_prefetch(W + (size_t)(k0 + TK + brow_ld) * N + n0 + bcol_ld, 0, 0);
#if USE_TDM
        __builtin_amdgcn_s_wait_tensorcnt(0);
#endif
        __syncthreads();

        // A fragment: two contiguous 16B runs per lane
        frag16 af;
        const int ar = w * 16 + nl;
        af.q[0] = *(const uint4*)&As[ar][8 * half];
        af.q[1] = *(const uint4*)&As[ar][16 + 8 * half];
#pragma unroll
        for (int t4 = 0; t4 < 4; ++t4) {
            // B fragment: k = 8*half..+7 and 16+8*half..+7 in BsT row n
            frag16 bf;
            const int bn = t4 * 16 + nl;
            bf.q[0] = *(const uint4*)&BsT[bn][8 * half];
            bf.q[1] = *(const uint4*)&BsT[bn][16 + 8 * half];
            acc[t4] = WMMA(af.v, bf.v, acc[t4]);
        }
    }

    // Epilogue: C element r -> row r + 8*half, col = lane&15
#pragma unroll
    for (int t4 = 0; t4 < 4; ++t4) {
#pragma unroll
        for (int r = 0; r < 8; ++r) {
            int row = m0 + w * 16 + r + 8 * half;
            int col = n0 + t4 * 16 + nl;
            float v = acc[t4][r] + (bias ? bias[col] : 0.f);
            if (outF) {
                outF[(size_t)row * N + col] = v;
            } else {
                out0[(size_t)row * N + col] = (h16)(v + (add0 ? add0[col] : 0.f));
                if (out1)
                    out1[(size_t)row * N + col] = (h16)(v + add1[col]);
            }
        }
    }
}

// ---------------------------------------------------------------------------
// Attention kernel: one wave32 per (b, h, 16-row q block).
// rel_shift(BD)[q,k] = BD[q, S-1-q+k]  (k<=q)
//                    = 0               (k==q+1)
//                    = BD[q+1, k-q-2]  (k>=q+2)
// BD kept transposed in LDS: BDT[j][a]  (a = q - q0, 0..16)
// ---------------------------------------------------------------------------
__global__ __launch_bounds__(32) void attn_kernel(
    const h16* __restrict__ Qu, const h16* __restrict__ Qv,
    const h16* __restrict__ Kh, const h16* __restrict__ Ph,
    const h16* __restrict__ Vh, h16* __restrict__ Ctx)
{
    __shared__ __align__(16) h16 BDT[S_LEN][24];     // 49152 B, 48B row stride
    __shared__ __align__(16) h16 VtT[DH][32 + 8];    // transposed V tile, 80B stride
    __shared__ __align__(16) h16 Pst[16][32 + 8];    // prob restage, 80B stride
    __shared__ float rowM[16], rowS[16];

    const int q0 = blockIdx.x * 16;
    const int h  = blockIdx.y;
    const int b  = blockIdx.z;
    const int lane = threadIdx.x;
    const int nl = lane & 15, half = lane >> 4;
    const size_t rowBase = (size_t)b * S_LEN;
    const int colOff = h * DH;

    v8f z = {};

    // ---- Phase A: BDT[j][a] = (Qv row q0+a) . (P row j) ----
    const h16* qvrow = Qv + (rowBase + q0 + nl) * DMODEL + colOff;
    frag16 avq[2];
#pragma unroll
    for (int c = 0; c < 2; ++c) {
        avq[c].q[0] = *(const uint4*)(qvrow + 32 * c + 16 * half);
        avq[c].q[1] = *(const uint4*)(qvrow + 32 * c + 16 * half + 8);
    }
    for (int j0 = 0; j0 < S_LEN; j0 += 16) {
        const h16* prow = Ph + (rowBase + j0 + nl) * DMODEL + colOff;
        v8f acc = z;
#pragma unroll
        for (int c = 0; c < 2; ++c) {
            v16h bf = *(const v16h*)(prow + 32 * c + 16 * half);  // 32B aligned
            acc = WMMA(avq[c].v, bf, acc);
        }
        pack8 pk;
#pragma unroll
        for (int r = 0; r < 8; ++r) pk.e[r] = (h16)acc[r];
        *(uint4*)&BDT[j0 + nl][8 * half] = pk.q;   // one 16B store per lane
    }
    // overlap row a=16 (wrap part of q row q0+15)
    if (q0 + 16 < S_LEN) {
        const h16* qv16 = Qv + (rowBase + q0 + 16) * DMODEL + colOff;
        for (int j = lane; j < S_LEN; j += 32) {
            const h16* pj = Ph + (rowBase + j) * DMODEL + colOff;
            float s = 0.f;
            for (int d = 0; d < DH; ++d)
                s += (float)qv16[d] * (float)pj[d];
            BDT[j][16] = (h16)s;
        }
    } else {
        for (int j = lane; j < S_LEN; j += 32) BDT[j][16] = (h16)0.f;
    }
    if (lane < 16) { rowM[lane] = -3.0e38f; rowS[lane] = 0.f; }
    __syncthreads();

    // Qu fragments (content scores)
    const h16* qurow = Qu + (rowBase + q0 + nl) * DMODEL + colOff;
    frag16 auq[2];
#pragma unroll
    for (int c = 0; c < 2; ++c) {
        auq[c].q[0] = *(const uint4*)(qurow + 32 * c + 16 * half);
        auq[c].q[1] = *(const uint4*)(qurow + 32 * c + 16 * half + 8);
    }

    auto scoreTile = [&](int j0s, float* sc) {
        const h16* krow = Kh + (rowBase + j0s + nl) * DMODEL + colOff;
        v8f acc = z;
#pragma unroll
        for (int c = 0; c < 2; ++c) {
            v16h bf = *(const v16h*)(krow + 32 * c + 16 * half);
            acc = WMMA(auq[c].v, bf, acc);
        }
        int k = j0s + nl;
#pragma unroll
        for (int r = 0; r < 8; ++r) {
            int a = r + 8 * half;
            int q = q0 + a;
            float pv;
            if (k <= q)          pv = (float)BDT[S_LEN - 1 - q + k][a];
            else if (k == q + 1) pv = 0.f;
            else                 pv = (float)BDT[k - q - 2][a + 1];
            sc[r] = (acc[r] + pv) * SCALE;
        }
    };

    // ---- Pass 1: online row max / sum ----
    for (int j0 = 0; j0 < S_LEN; j0 += 16) {
        float sc[8];
        scoreTile(j0, sc);
        float tmax[8], tsum[8];
#pragma unroll
        for (int r = 0; r < 8; ++r) {
            float v = sc[r];
            for (int m = 8; m >= 1; m >>= 1) v = fmaxf(v, __shfl_xor(v, m, 32));
            tmax[r] = v;
            float e = __expf(sc[r] - v);
            for (int m = 8; m >= 1; m >>= 1) e += __shfl_xor(e, m, 32);
            tsum[r] = e;
        }
        if (nl == 0) {   // lanes 0 and 16 update disjoint rows
#pragma unroll
            for (int r = 0; r < 8; ++r) {
                int idx = r + 8 * half;
                float mo = rowM[idx];
                float mn = fmaxf(mo, tmax[r]);
                rowS[idx] = rowS[idx] * __expf(mo - mn) + tsum[r] * __expf(tmax[r] - mn);
                rowM[idx] = mn;
            }
        }
    }
    __syncthreads();

    // ---- Pass 2: probabilities -> context via WMMA ----
    v8f cacc[4];
#pragma unroll
    for (int i = 0; i < 4; ++i) cacc[i] = z;

    for (int j0 = 0; j0 < S_LEN; j0 += 32) {
        __syncthreads();
        {   // stage V rows j0..j0+31 transposed: VtT[d][kk]
            pack64 vv;
            const uint4* s = (const uint4*)(Vh + (rowBase + j0 + lane) * DMODEL + colOff);
#pragma unroll
            for (int i = 0; i < 8; ++i) vv.q[i] = s[i];
#pragma unroll
            for (int d = 0; d < DH; ++d) VtT[d][lane] = vv.e[d];
        }
        __syncthreads();
#pragma unroll
        for (int sub = 0; sub < 2; ++sub) {
            float sc[8];
            scoreTile(j0 + sub * 16, sc);
#pragma unroll
            for (int r = 0; r < 8; ++r) {
                int idx = r + 8 * half;
                float p = __expf(sc[r] - rowM[idx]) / rowS[idx];
                Pst[idx][sub * 16 + nl] = (h16)p;
            }
        }
        __syncthreads();
        // probabilities in A layout: two contiguous 16B runs
        frag16 pf;
        pf.q[0] = *(const uint4*)&Pst[nl][8 * half];
        pf.q[1] = *(const uint4*)&Pst[nl][16 + 8 * half];
#pragma unroll
        for (int t4 = 0; t4 < 4; ++t4) {
            frag16 vf;
            const int vn = t4 * 16 + nl;
            vf.q[0] = *(const uint4*)&VtT[vn][8 * half];
            vf.q[1] = *(const uint4*)&VtT[vn][16 + 8 * half];
            cacc[t4] = WMMA(pf.v, vf.v, cacc[t4]);
        }
    }

#pragma unroll
    for (int t4 = 0; t4 < 4; ++t4)
#pragma unroll
        for (int r = 0; r < 8; ++r) {
            int q = q0 + r + 8 * half;
            Ctx[(rowBase + q) * DMODEL + colOff + t4 * 16 + nl] = (h16)cacc[t4][r];
        }
}

// ---------------------------------------------------------------------------
extern "C" void kernel_launch(void* const* d_in, const int* in_sizes, int n_in,
                              void* d_out, int out_size, void* d_ws, size_t ws_size,
                              hipStream_t stream)
{
    const float* query = (const float*)d_in[0];
    const float* key   = (const float*)d_in[1];
    const float* value = (const float*)d_in[2];
    const float* pemb  = (const float*)d_in[3];
    const float* Wq  = (const float*)d_in[4];
    const float* bq  = (const float*)d_in[5];
    const float* Wk  = (const float*)d_in[6];
    const float* bk  = (const float*)d_in[7];
    const float* Wv  = (const float*)d_in[8];
    const float* bv  = (const float*)d_in[9];
    const float* Wp  = (const float*)d_in[10];
    const float* Wo  = (const float*)d_in[11];
    const float* bo  = (const float*)d_in[12];
    const float* ub  = (const float*)d_in[13];  // [H,dh] -> flat 512 per-column add
    const float* vb  = (const float*)d_in[14];
    float* out = (float*)d_out;

    const size_t AE = (size_t)NROWS * DMODEL;   // 4,194,304
    const size_t WE = (size_t)DMODEL * DMODEL;  // 262,144

    h16* ws   = (h16*)d_ws;
    h16* q16  = ws; ws += AE;
    h16* k16  = ws; ws += AE;
    h16* v16  = ws; ws += AE;
    h16* p16  = ws; ws += AE;
    h16* wq16 = ws; ws += WE;
    h16* wk16 = ws; ws += WE;
    h16* wv16 = ws; ws += WE;
    h16* wp16 = ws; ws += WE;
    h16* wo16 = ws; ws += WE;
    h16* Qu   = ws; ws += AE;
    h16* Qv   = ws; ws += AE;
    h16* Kh   = ws; ws += AE;
    h16* Ph   = ws; ws += AE;
    h16* Vh   = ws; ws += AE;
    h16* Ctx  = ws; ws += AE;
    // total ~86.5 MB of workspace

    const int CB = 256;
    f32_to_f16_kernel<<<(int)(AE / 4 / CB), CB, 0, stream>>>(query, q16, (int)(AE / 4));
    f32_to_f16_kernel<<<(int)(AE / 4 / CB), CB, 0, stream>>>(key,   k16, (int)(AE / 4));
    f32_to_f16_kernel<<<(int)(AE / 4 / CB), CB, 0, stream>>>(value, v16, (int)(AE / 4));
    f32_to_f16_kernel<<<(int)(AE / 4 / CB), CB, 0, stream>>>(pemb,  p16, (int)(AE / 4));
    f32_to_f16_kernel<<<(int)(WE / 4 / CB), CB, 0, stream>>>(Wq, wq16, (int)(WE / 4));
    f32_to_f16_kernel<<<(int)(WE / 4 / CB), CB, 0, stream>>>(Wk, wk16, (int)(WE / 4));
    f32_to_f16_kernel<<<(int)(WE / 4 / CB), CB, 0, stream>>>(Wv, wv16, (int)(WE / 4));
    f32_to_f16_kernel<<<(int)(WE / 4 / CB), CB, 0, stream>>>(Wp, wp16, (int)(WE / 4));
    f32_to_f16_kernel<<<(int)(WE / 4 / CB), CB, 0, stream>>>(Wo, wo16, (int)(WE / 4));

    dim3 gg(DMODEL / TN, NROWS / TM);   // (8, 64)
    gemm16_kernel<<<gg, 256, 0, stream>>>(q16, wq16, bq, ub, vb, Qu, Qv, nullptr, NROWS, DMODEL, DMODEL);
    gemm16_kernel<<<gg, 256, 0, stream>>>(k16, wk16, bk, nullptr, nullptr, Kh, nullptr, nullptr, NROWS, DMODEL, DMODEL);
    gemm16_kernel<<<gg, 256, 0, stream>>>(v16, wv16, bv, nullptr, nullptr, Vh, nullptr, nullptr, NROWS, DMODEL, DMODEL);
    gemm16_kernel<<<gg, 256, 0, stream>>>(p16, wp16, nullptr, nullptr, nullptr, Ph, nullptr, nullptr, NROWS, DMODEL, DMODEL);

    dim3 ga(S_LEN / 16, 8, 8);          // (64, H, B)
    attn_kernel<<<ga, 32, 0, stream>>>(Qu, Qv, Kh, Ph, Vh, Ctx);

    gemm16_kernel<<<gg, 256, 0, stream>>>(Ctx, wo16, bo, nullptr, nullptr, nullptr, nullptr, out, NROWS, DMODEL, DMODEL);
}
